// MultiHead2_34961033790133
// MI455X (gfx1250) — compile-verified, matrix operand order checked by reading, not run on hardware
//
#include <hip/hip_runtime.h>

// ---------------------------------------------------------------------------
// Problem constants (from reference): B=4, T=1024, D=1024, N=16, DH=64
// ---------------------------------------------------------------------------
#define Bc   4
#define Tc   1024
#define Dc   1024
#define NHc  16
#define DHc  64
#define INFV 1.0e10f
#define INV_SCALE (1.0f / 32.0f)   // 1/sqrt(D), D=1024

typedef __attribute__((ext_vector_type(16))) __bf16 v16bf;
typedef __attribute__((ext_vector_type(8)))  __bf16 v8bf;
typedef __attribute__((ext_vector_type(4)))  __bf16 v4bf;
typedef __attribute__((ext_vector_type(8)))  float  v8f;
typedef __attribute__((ext_vector_type(4)))  float  v4f;

// Load a 16x32 bf16 A/B fragment's per-lane data: two contiguous 16B chunks.
// p must already include the lane's (hi*8) K offset; chunk2 is +16 halves.
__device__ __forceinline__ v16bf load_frag(const __bf16* p) {
    v8bf a = *(const v8bf*)p;
    v8bf b = *(const v8bf*)(p + 16);
    return __builtin_shufflevector(a, b, 0, 1, 2, 3, 4, 5, 6, 7,
                                          8, 9, 10, 11, 12, 13, 14, 15);
}

__device__ __forceinline__ v8f wmma_bf16(v16bf a, v16bf b, v8f c) {
    return __builtin_amdgcn_wmma_f32_16x16x32_bf16(
        /*neg_a=*/false, a, /*neg_b=*/false, b,
        /*c_mod=*/(short)0, c, /*reuse_a=*/false, /*reuse_b=*/false);
}

// ---------------------------------------------------------------------------
// f32 -> bf16 conversion (vectorized x4)
// ---------------------------------------------------------------------------
__global__ __launch_bounds__(256) void cvt_kernel(const float* __restrict__ in,
                                                  __bf16* __restrict__ out,
                                                  int n4) {
    int i = blockIdx.x * blockDim.x + threadIdx.x;
    if (i < n4) {
        v4f x = ((const v4f*)in)[i];
        v4bf y;
        y[0] = (__bf16)x[0];
        y[1] = (__bf16)x[1];
        y[2] = (__bf16)x[2];
        y[3] = (__bf16)x[3];
        ((v4bf*)out)[i] = y;
    }
}

// ---------------------------------------------------------------------------
// GEMM: out = A (M x K, bf16 rowmajor) * W^T (W is N x K bf16 rowmajor) + bias
// Each wave computes a 16(M) x 64(N) tile: 1 A-fragment reused by 4 WMMAs/step.
// MODE 0: write bf16 out [M,N].  MODE 1: write f32 out [M,N].
// MODE 2: write bf16 head-transposed vt[((b*NH+h)*DH+dh)*T + t].
// M = B*T = 4096, N = K = D = 1024.
// ---------------------------------------------------------------------------
template <int MODE>
__global__ __launch_bounds__(256) void gemm_kernel(
    const __bf16* __restrict__ A, const __bf16* __restrict__ W,
    const float* __restrict__ bias, __bf16* __restrict__ outb,
    float* __restrict__ outf, __bf16* __restrict__ vt) {
    const int wave = threadIdx.x >> 5;
    const int lane = threadIdx.x & 31;
    const int hi = lane >> 4, lo = lane & 15;
    const int gid = blockIdx.x * 8 + wave;   // 4096 wave-tiles
    const int tm = gid >> 4;                 // 256 M-tiles (16 rows each)
    const int tn = gid & 15;                 // 16 N-chunks (64 cols each)
    const int row0 = tm * 16, col0 = tn * 64;

    v8f c0 = {}, c1 = {}, c2 = {}, c3 = {};
    const __bf16* arow = A + ((size_t)(row0 + lo)) * Dc + hi * 8;
    const __bf16* w0   = W + ((size_t)(col0 + lo)) * Dc + hi * 8;

#pragma unroll 4
    for (int kk = 0; kk < Dc; kk += 32) {
        v16bf af = load_frag(arow + kk);
        c0 = wmma_bf16(af, load_frag(w0 + kk), c0);
        c1 = wmma_bf16(af, load_frag(w0 + (size_t)16 * Dc + kk), c1);
        c2 = wmma_bf16(af, load_frag(w0 + (size_t)32 * Dc + kk), c2);
        c3 = wmma_bf16(af, load_frag(w0 + (size_t)48 * Dc + kk), c3);
    }

    float bv0 = bias[col0 + lo];
    float bv1 = bias[col0 + 16 + lo];
    float bv2 = bias[col0 + 32 + lo];
    float bv3 = bias[col0 + 48 + lo];

#pragma unroll
    for (int i = 0; i < 8; ++i) {
        const int row = row0 + i + 8 * hi;
        float r0 = c0[i] + bv0, r1 = c1[i] + bv1;
        float r2 = c2[i] + bv2, r3 = c3[i] + bv3;
        if constexpr (MODE == 1) {
            float* p = outf + (size_t)row * Dc + col0 + lo;
            p[0] = r0; p[16] = r1; p[32] = r2; p[48] = r3;
        } else if constexpr (MODE == 0) {
            __bf16* p = outb + (size_t)row * Dc + col0 + lo;
            p[0] = (__bf16)r0; p[16] = (__bf16)r1;
            p[32] = (__bf16)r2; p[48] = (__bf16)r3;
        } else {
            const int bb = row >> 10, t = row & (Tc - 1);
#pragma unroll
            for (int j = 0; j < 4; ++j) {
                const int col = col0 + j * 16 + lo;
                const int hh = col >> 6, dh = col & (DHc - 1);
                float r = (j == 0) ? r0 : (j == 1) ? r1 : (j == 2) ? r2 : r3;
                vt[(((size_t)(bb * NHc + hh)) * DHc + dh) * Tc + t] = (__bf16)r;
            }
        }
    }
}

// ---------------------------------------------------------------------------
// Flash attention: block = 4 waves x 32 lanes; each wave owns 16 query rows
// of one (b, h). Online softmax in f32; P staged via LDS to reshape the
// WMMA C-layout into an A-fragment.
// ---------------------------------------------------------------------------
__global__ __launch_bounds__(128) void attn_kernel(
    const __bf16* __restrict__ Q, const __bf16* __restrict__ Kb,
    const __bf16* __restrict__ Vt, const float* __restrict__ wts,
    const float* __restrict__ mask, __bf16* __restrict__ O) {
    __shared__ __align__(16) __bf16 pbuf[4][16][32];

    const int wave = threadIdx.x >> 5;
    const int lane = threadIdx.x & 31;
    const int hi = lane >> 4, lo = lane & 15;
    const int qblk = blockIdx.x & (Tc / 64 - 1);  // 16 q-blocks of 64
    const int bh = blockIdx.x >> 4;               // b*NH + h
    const int h = bh & (NHc - 1), b = bh >> 4;
    const int q0 = qblk * 64 + wave * 16;

    // Q A-fragments for this wave's 16 rows (K-dim = DH = 64 -> 2 fragments)
    const __bf16* qbase = Q + ((size_t)(b * Tc + q0 + lo)) * Dc + h * DHc + hi * 8;
    v16bf qa0 = load_frag(qbase);
    v16bf qa1 = load_frag(qbase + 32);

    v8f o0 = {}, o1 = {}, o2 = {}, o3 = {};
    float mrow[8], lrow[8];
#pragma unroll
    for (int i = 0; i < 8; ++i) { mrow[i] = -1e30f; lrow[i] = 0.0f; }

    for (int j0 = 0; j0 <= q0 + 15; j0 += 32) {
        // ---- S = Q * K^T for a 16x32 key block (two 16x16 C tiles) ----
        v8f s0 = {}, s1 = {};
        {
            const __bf16* kb0 = Kb + ((size_t)(b * Tc + j0 + lo)) * Dc + h * DHc + hi * 8;
            const __bf16* kb1 = kb0 + (size_t)16 * Dc;
            s0 = wmma_bf16(qa0, load_frag(kb0), s0);
            s0 = wmma_bf16(qa1, load_frag(kb0 + 32), s0);
            s1 = wmma_bf16(qa0, load_frag(kb1), s1);
            s1 = wmma_bf16(qa1, load_frag(kb1 + 32), s1);
        }

        // ---- bias + causal + padding, online softmax ----
        float p0[8], p1[8];
#pragma unroll
        for (int i = 0; i < 8; ++i) {
            const int qt = q0 + i + 8 * hi;
            const int k0 = j0 + lo;
            const int k1 = j0 + 16 + lo;
            const float* wr = wts + ((size_t)bh * Tc + qt) * Tc;
            __builtin_prefetch(wr + k0 + 32, 0, 1);
            float u0 = s0[i] + wr[k0];
            float u1 = s1[i] + wr[k1];
            u0 -= (k0 > qt) ? INFV : 0.0f;
            u1 -= (k1 > qt) ? INFV : 0.0f;
            u0 -= (1.0f - mask[b * Tc + k0]) * INFV;
            u1 -= (1.0f - mask[b * Tc + k1]) * INFV;
            u0 *= INV_SCALE;
            u1 *= INV_SCALE;
            // row max across the 16 lanes holding this row
            float ml = fmaxf(u0, u1);
            ml = fmaxf(ml, __shfl_xor(ml, 1, 16));
            ml = fmaxf(ml, __shfl_xor(ml, 2, 16));
            ml = fmaxf(ml, __shfl_xor(ml, 4, 16));
            ml = fmaxf(ml, __shfl_xor(ml, 8, 16));
            float mnew = fmaxf(mrow[i], ml);
            float cf = __expf(mrow[i] - mnew);
            float e0 = __expf(u0 - mnew);
            float e1 = __expf(u1 - mnew);
            float rs = e0 + e1;
            rs += __shfl_xor(rs, 1, 16);
            rs += __shfl_xor(rs, 2, 16);
            rs += __shfl_xor(rs, 4, 16);
            rs += __shfl_xor(rs, 8, 16);
            lrow[i] = lrow[i] * cf + rs;
            mrow[i] = mnew;
            o0[i] *= cf; o1[i] *= cf; o2[i] *= cf; o3[i] *= cf;
            p0[i] = e0; p1[i] = e1;
        }

        // ---- stage P to LDS: C-layout -> A-fragment layout ----
#pragma unroll
        for (int i = 0; i < 8; ++i) {
            pbuf[wave][i + 8 * hi][lo]      = (__bf16)p0[i];
            pbuf[wave][i + 8 * hi][16 + lo] = (__bf16)p1[i];
        }
        asm volatile("s_wait_dscnt 0" ::: "memory");
        v16bf pa = load_frag(&pbuf[wave][lo][hi * 8]);

        // ---- O += P * V  (Vt is head-major: rows contiguous over t) ----
        const __bf16* vb = Vt + ((size_t)bh * DHc + lo) * Tc + j0 + hi * 8;
        o0 = wmma_bf16(pa, load_frag(vb), o0);
        o1 = wmma_bf16(pa, load_frag(vb + (size_t)16 * Tc), o1);
        o2 = wmma_bf16(pa, load_frag(vb + (size_t)32 * Tc), o2);
        o3 = wmma_bf16(pa, load_frag(vb + (size_t)48 * Tc), o3);
    }

    // ---- normalize + write merged-head bf16 output ----
#pragma unroll
    for (int i = 0; i < 8; ++i) {
        const float rl = 1.0f / lrow[i];
        const size_t row = (size_t)(b * Tc + q0 + i + 8 * hi);
        __bf16* orow = O + row * Dc + h * DHc + lo;
        orow[0]  = (__bf16)(o0[i] * rl);
        orow[16] = (__bf16)(o1[i] * rl);
        orow[32] = (__bf16)(o2[i] * rl);
        orow[48] = (__bf16)(o3[i] * rl);
    }
}

// ---------------------------------------------------------------------------
// Host-side orchestration
// ---------------------------------------------------------------------------
extern "C" void kernel_launch(void* const* d_in, const int* in_sizes, int n_in,
                              void* d_out, int out_size, void* d_ws, size_t ws_size,
                              hipStream_t stream) {
    (void)in_sizes; (void)n_in; (void)out_size; (void)ws_size;

    const float* q    = (const float*)d_in[0];
    const float* k    = (const float*)d_in[1];
    const float* v    = (const float*)d_in[2];
    const float* mask = (const float*)d_in[3];
    const float* wts  = (const float*)d_in[4];
    const float* Wq   = (const float*)d_in[5];
    const float* bq   = (const float*)d_in[6];
    const float* Wk   = (const float*)d_in[7];
    const float* bk   = (const float*)d_in[8];
    const float* Wv   = (const float*)d_in[9];
    const float* bv   = (const float*)d_in[10];
    const float* Wo   = (const float*)d_in[11];
    const float* bo   = (const float*)d_in[12];
    float* out = (float*)d_out;

    char* ws = (char*)d_ws;
    size_t off = 0;
    auto alloc = [&](size_t bytes) -> char* {
        char* p = ws + off;
        off += (bytes + 255) & ~(size_t)255;
        return p;
    };
    const size_t actBytes = (size_t)Bc * Tc * Dc * 2;  // 8 MB
    const size_t wBytes   = (size_t)Dc * Dc * 2;       // 2 MB

    __bf16* Aq  = (__bf16*)alloc(actBytes);
    __bf16* Ak  = (__bf16*)alloc(actBytes);
    __bf16* Av  = (__bf16*)alloc(actBytes);
    __bf16* Wqb = (__bf16*)alloc(wBytes);
    __bf16* Wkb = (__bf16*)alloc(wBytes);
    __bf16* Wvb = (__bf16*)alloc(wBytes);
    __bf16* Wob = (__bf16*)alloc(wBytes);
    __bf16* Qb  = (__bf16*)alloc(actBytes);
    __bf16* Kbf = (__bf16*)alloc(actBytes);
    __bf16* Vt  = (__bf16*)alloc(actBytes);
    __bf16* Obf = Aq;  // reuse: query-bf16 dead after the Q projection

    const int nAct4 = Bc * Tc * Dc / 4;  // 1M vec4
    const int nW4   = Dc * Dc / 4;       // 256K vec4

    cvt_kernel<<<nAct4 / 256, 256, 0, stream>>>(q, Aq, nAct4);
    cvt_kernel<<<nAct4 / 256, 256, 0, stream>>>(k, Ak, nAct4);
    cvt_kernel<<<nAct4 / 256, 256, 0, stream>>>(v, Av, nAct4);
    cvt_kernel<<<nW4 / 256, 256, 0, stream>>>(Wq, Wqb, nW4);
    cvt_kernel<<<nW4 / 256, 256, 0, stream>>>(Wk, Wkb, nW4);
    cvt_kernel<<<nW4 / 256, 256, 0, stream>>>(Wv, Wvb, nW4);
    cvt_kernel<<<nW4 / 256, 256, 0, stream>>>(Wo, Wob, nW4);

    // 4096 wave-tiles / 8 waves per block = 512 blocks
    gemm_kernel<0><<<512, 256, 0, stream>>>(Aq, Wqb, bq, Qb, nullptr, nullptr);
    gemm_kernel<0><<<512, 256, 0, stream>>>(Ak, Wkb, bk, Kbf, nullptr, nullptr);
    gemm_kernel<2><<<512, 256, 0, stream>>>(Av, Wvb, bv, nullptr, nullptr, Vt);

    // B*N*(T/64) = 1024 blocks, 4 waves each
    attn_kernel<<<Bc * NHc * (Tc / 64), 128, 0, stream>>>(Qb, Kbf, Vt, wts, mask, Obf);

    gemm_kernel<1><<<512, 256, 0, stream>>>(Obf, Wob, bo, nullptr, out, nullptr);
}
